// RegionGrouping_30382598652306
// MI455X (gfx1250) — compile-verified, matrix operand order checked by reading, not run on hardware
//
#include <hip/hip_runtime.h>

// ---------------------------------------------------------------------------
// RegionGrouping for MI455X (gfx1250, wave32, WMMA bf16 16x16x32, f32 accum)
// ---------------------------------------------------------------------------

typedef __attribute__((ext_vector_type(16))) __bf16    v16bf;
typedef __attribute__((ext_vector_type(8)))  float     v8f;
typedef __attribute__((ext_vector_type(8)))  unsigned  v8u;
typedef int v4i_g __attribute__((__vector_size__(4 * sizeof(int))));  // matches builtin proto

#define NREG 16
#define BM   128
#define BN   128
#define BK   32
#define LDA_U 20     // uint stride per A-tile row: 16 data + 4 pad (16B-aligned rows)
#define LDB_U 132    // uint stride per B pair-row (BN + 4 pad)

#if defined(__has_builtin)
#if __has_builtin(__builtin_amdgcn_global_load_async_to_lds_b128)
#define HAVE_ASYNC_LDS 1
#endif
#if __has_builtin(__builtin_amdgcn_s_wait_asynccnt)
#define HAVE_WAIT_ASYNC_BUILTIN 1
#endif
#endif

__device__ __forceinline__ void wait_async_zero() {
#if defined(HAVE_ASYNC_LDS)
#if defined(HAVE_WAIT_ASYNC_BUILTIN)
  __builtin_amdgcn_s_wait_asynccnt(0);
#else
  asm volatile("s_wait_asynccnt 0x0" ::: "memory");
#endif
#endif
}

__device__ __forceinline__ unsigned short f2bf(float f) {
  unsigned u = __float_as_uint(f);
  u += 0x7FFFu + ((u >> 16) & 1u);            // round-to-nearest-even
  return (unsigned short)(u >> 16);
}
__device__ __forceinline__ unsigned pack2(float lo, float hi) {
  return (unsigned)f2bf(lo) | ((unsigned)f2bf(hi) << 16);
}

// kp(j,half): uint (bf16-pair) index within a 32-wide K slice, per ISA 7.12.2
__device__ __forceinline__ int kpair(int j, int hh) {
  return (j & 3) + ((j >> 2) << 3) + (hh << 2);
}

// Load one A fragment (16x32 bf16) for tile row base `row`: two b128 LDS reads.
__device__ __forceinline__ v8u load_afrag(const unsigned* ldsA, int row, int hh) {
  const uint4 lo = *reinterpret_cast<const uint4*>(&ldsA[row * LDA_U + (hh << 2)]);
  const uint4 hi = *reinterpret_cast<const uint4*>(&ldsA[row * LDA_U + 8 + (hh << 2)]);
  v8u a;
  a[0] = lo.x; a[1] = lo.y; a[2] = lo.z; a[3] = lo.w;
  a[4] = hi.x; a[5] = hi.y; a[6] = hi.z; a[7] = hi.w;
  return a;
}

// --------------------------- 1) occ / top-1 --------------------------------
__global__ __launch_bounds__(256) void occ_kernel(
    const float* __restrict__ x, const float* __restrict__ occ_w,
    const float* __restrict__ occ_b, int* __restrict__ reg_indx,
    float* __restrict__ reg_logit, int S, int D) {
  const int row  = blockIdx.x * 8 + (threadIdx.x >> 5);
  const int lane = threadIdx.x & 31;
  float acc[NREG];
#pragma unroll
  for (int r = 0; r < NREG; ++r) acc[r] = 0.f;

  const float* xr = x + (size_t)row * D;
  for (int k = lane; k < D; k += 32) {
    float xv = xr[k];
    const float* wk = occ_w + (size_t)k * NREG;
#pragma unroll
    for (int r = 0; r < NREG; ++r) acc[r] += xv * wk[r];
  }
#pragma unroll
  for (int off = 16; off > 0; off >>= 1) {
#pragma unroll
    for (int r = 0; r < NREG; ++r) acc[r] += __shfl_xor(acc[r], off, 32);
  }
  if (lane == 0) {
    float lg[NREG];
    float m = -3.4e38f; int mi = 0;
#pragma unroll
    for (int r = 0; r < NREG; ++r) {
      lg[r] = acc[r] + occ_b[r];
      if (lg[r] > m) { m = lg[r]; mi = r; }   // strict > keeps first max (argmax tie rule)
    }
    float se = 0.f;
#pragma unroll
    for (int r = 0; r < NREG; ++r) se += __expf(lg[r] - m);
    reg_indx[row]  = mi;
    reg_logit[row] = 1.f / se;                 // softmax prob of the max logit
  }
}

// ------------------- 2) deterministic per-(b,r) stats ----------------------
__global__ void stats_kernel(const int* __restrict__ reg_indx,
                             const float* __restrict__ reg_logit,
                             float* __restrict__ sumlogit, int* __restrict__ counts,
                             int S) {
  const int t = threadIdx.x;                   // 0..B*NREG-1 (128)
  const int b = t >> 4, rr = t & 15;
  const int*   ip = reg_indx  + (size_t)b * S;
  const float* lp = reg_logit + (size_t)b * S;
  float s = 0.f; int c = 0;
  for (int i = 0; i < S; ++i) {
    if (ip[i] == rr) { s += lp[i]; ++c; }
  }
  sumlogit[t] = s;
  counts[t]   = c;
}

// --------------------------- 3) GEMM1 (WMMA) -------------------------------
// h1 = relu(x @ w1 + b1), f32 in -> bf16 WMMA -> bf16 out
__global__ __launch_bounds__(256) void gemm1_kernel(
    const float* __restrict__ x, const float* __restrict__ w1,
    const float* __restrict__ b1, unsigned short* __restrict__ h1,
    int K, int N) {
  __shared__ unsigned ldsA[BM * LDA_U];        // bf16 pairs, padded, 16B-aligned rows
  __shared__ unsigned ldsB[(BK / 2) * LDB_U];

  const int tid = threadIdx.x;
  const int m0 = blockIdx.y * BM, n0 = blockIdx.x * BN;
  const int wave = tid >> 5, lane = tid & 31;
  const int r = lane & 15, hh = lane >> 4;
  const int wm = wave & 1, wn = wave >> 1;     // 2x4 wave grid, wave tile 64x32

  v8f acc[4][2] = {};

  for (int k0 = 0; k0 < K; k0 += BK) {
    // stage A tile 128x32 f32 -> bf16 pairs
#pragma unroll
    for (int i = 0; i < 8; ++i) {
      int e = tid + 256 * i;                   // 2048 pair entries
      int row = e >> 4, kpi = e & 15;
      const float2 xv = *(const float2*)(x + (size_t)(m0 + row) * K + k0 + kpi * 2);
      ldsA[row * LDA_U + kpi] = pack2(xv.x, xv.y);
    }
    // stage B tile 32x128 f32 -> pair-interleaved [16][128]
#pragma unroll
    for (int i = 0; i < 8; ++i) {
      int e = tid + 256 * i;
      int kpi = e >> 7, c = e & 127;
      const float* gp = w1 + (size_t)(k0 + kpi * 2) * N + n0 + c;
      ldsB[kpi * LDB_U + c] = pack2(gp[0], gp[N]);
    }
    if (k0 + BK < K)                           // prefetch next weight slab
      __builtin_prefetch(w1 + (size_t)(k0 + BK + ((tid >> 7) << 1)) * N + n0 + (tid & 127), 0, 0);
    __syncthreads();

    v8u a_u[4], b_u[2];
#pragma unroll
    for (int mi = 0; mi < 4; ++mi)
      a_u[mi] = load_afrag(ldsA, wm * 64 + mi * 16 + r, hh);
#pragma unroll
    for (int ni = 0; ni < 2; ++ni) {
      int col = wn * 32 + ni * 16 + r;
#pragma unroll
      for (int j = 0; j < 8; ++j) b_u[ni][j] = ldsB[kpair(j, hh) * LDB_U + col];
    }
#pragma unroll
    for (int mi = 0; mi < 4; ++mi)
#pragma unroll
      for (int ni = 0; ni < 2; ++ni)
        acc[mi][ni] = __builtin_amdgcn_wmma_f32_16x16x32_bf16(
            false, __builtin_bit_cast(v16bf, a_u[mi]),
            false, __builtin_bit_cast(v16bf, b_u[ni]),
            (short)0, acc[mi][ni], false, false);
    __syncthreads();
  }

  // epilogue: bias + relu -> bf16 store
#pragma unroll
  for (int mi = 0; mi < 4; ++mi) {
#pragma unroll
    for (int ni = 0; ni < 2; ++ni) {
      const int col = n0 + wn * 32 + ni * 16 + r;
      const float bias = b1[col];
#pragma unroll
      for (int v = 0; v < 8; ++v) {
        const int row = m0 + wm * 64 + mi * 16 + v + 8 * hh;
        float f = acc[mi][ni][v] + bias;
        f = f > 0.f ? f : 0.f;
        h1[(size_t)row * N + col] = f2bf(f);
      }
    }
  }
}

// --------------------------- 4) GEMM2 (WMMA) + region max -----------------
// h2 = relu(h1 @ w2 + b2); fused per-(batch, region, channel) max reduce.
// A tile (already bf16) is staged with GLOBAL_LOAD_ASYNC_TO_LDS_B128 when the
// toolchain exposes the builtin (ASYNCcnt path), else a sync b128 copy.
__global__ __launch_bounds__(256) void gemm2_kernel(
    const unsigned short* __restrict__ h1, const float* __restrict__ w2,
    const float* __restrict__ b2, const int* __restrict__ reg_indx,
    unsigned* __restrict__ regmax, int K, int N, int S) {
  __shared__ unsigned ldsA[BM * LDA_U];
  __shared__ unsigned ldsB[(BK / 2) * LDB_U];
  __shared__ int      s_idx[BM];
  __shared__ unsigned s_rmax[NREG * BN];

  const int tid = threadIdx.x;
  const int m0 = blockIdx.y * BM, n0 = blockIdx.x * BN;
  const int wave = tid >> 5, lane = tid & 31;
  const int r = lane & 15, hh = lane >> 4;
  const int wm = wave & 1, wn = wave >> 1;
  const int b_idx = m0 / S;                    // BM divides S -> constant per block

  if (tid < BM) s_idx[tid] = reg_indx[m0 + tid];
#pragma unroll
  for (int i = 0; i < 8; ++i) s_rmax[tid + 256 * i] = 0u;

  v8f acc[4][2] = {};
  const unsigned* h1u = (const unsigned*)h1;

  for (int k0 = 0; k0 < K; k0 += BK) {
    // A tile: raw bf16 copy, 16B granules (512 quarter-rows, 2 per thread)
#pragma unroll
    for (int i = 0; i < 2; ++i) {
      int e = tid + 256 * i;
      int row = e >> 2, q = e & 3;
      const unsigned* gsrc = h1u + (((size_t)(m0 + row) * K + k0) >> 1) + q * 4;
      unsigned* ldst = &ldsA[row * LDA_U + q * 4];
#if defined(HAVE_ASYNC_LDS)
      __builtin_amdgcn_global_load_async_to_lds_b128(
          (__attribute__((address_space(1))) v4i_g*)gsrc,
          (__attribute__((address_space(3))) v4i_g*)ldst, 0, 0);
#else
      *reinterpret_cast<uint4*>(ldst) = *reinterpret_cast<const uint4*>(gsrc);
#endif
    }
    // B tile: f32 -> bf16 pair-interleaved [16][128]
#pragma unroll
    for (int i = 0; i < 8; ++i) {
      int e = tid + 256 * i;
      int kpi = e >> 7, c = e & 127;
      const float* gp = w2 + (size_t)(k0 + kpi * 2) * N + n0 + c;
      ldsB[kpi * LDB_U + c] = pack2(gp[0], gp[N]);
    }
    if (k0 + BK < K)
      __builtin_prefetch(w2 + (size_t)(k0 + BK + ((tid >> 7) << 1)) * N + n0 + (tid & 127), 0, 0);
    wait_async_zero();                         // drain ASYNCcnt before barrier
    __syncthreads();

    v8u a_u[4], b_u[2];
#pragma unroll
    for (int mi = 0; mi < 4; ++mi)
      a_u[mi] = load_afrag(ldsA, wm * 64 + mi * 16 + r, hh);
#pragma unroll
    for (int ni = 0; ni < 2; ++ni) {
      int col = wn * 32 + ni * 16 + r;
#pragma unroll
      for (int j = 0; j < 8; ++j) b_u[ni][j] = ldsB[kpair(j, hh) * LDB_U + col];
    }
#pragma unroll
    for (int mi = 0; mi < 4; ++mi)
#pragma unroll
      for (int ni = 0; ni < 2; ++ni)
        acc[mi][ni] = __builtin_amdgcn_wmma_f32_16x16x32_bf16(
            false, __builtin_bit_cast(v16bf, a_u[mi]),
            false, __builtin_bit_cast(v16bf, b_u[ni]),
            (short)0, acc[mi][ni], false, false);
    __syncthreads();
  }

  // epilogue: relu then region-wise max into LDS (ReLU => >=0 so uint-bit max == float max)
#pragma unroll
  for (int mi = 0; mi < 4; ++mi) {
#pragma unroll
    for (int ni = 0; ni < 2; ++ni) {
      const int cloc = wn * 32 + ni * 16 + r;
      const float bias = b2[n0 + cloc];
#pragma unroll
      for (int v = 0; v < 8; ++v) {
        const int rloc = wm * 64 + mi * 16 + v + 8 * hh;
        float f = acc[mi][ni][v] + bias;
        f = f > 0.f ? f : 0.f;
        unsigned u = __float_as_uint(f);
        if (u) atomicMax(&s_rmax[s_idx[rloc] * BN + cloc], u);
      }
    }
  }
  __syncthreads();
  // flush tile maxima to global (atomicMax is order-independent => deterministic)
#pragma unroll
  for (int i = 0; i < 8; ++i) {
    int e = tid + 256 * i;                     // NREG*BN = 2048 entries
    int rr = e >> 7, c = e & 127;
    unsigned u = s_rmax[e];
    if (u) atomicMax(&regmax[((size_t)(b_idx * NREG + rr)) * N + n0 + c], u);
  }
}

// --------------------------- 5) h_const = relu(relu(b1)@w2 + b2) -----------
__global__ void hconst_kernel(const float* __restrict__ b1v,
                              const float* __restrict__ w2,
                              const float* __restrict__ b2v,
                              float* __restrict__ h2c, int D) {
  const int d = blockIdx.x * 256 + threadIdx.x;
  if (d >= D) return;
  float acc = b2v[d];
  for (int k = 0; k < D; ++k) {
    float h = b1v[k];
    h = h > 0.f ? h : 0.f;
    acc += h * w2[(size_t)k * D + d];
  }
  h2c[d] = acc > 0.f ? acc : 0.f;
}

// --------------------------- 6) merge h_const into region max --------------
__global__ void merge_kernel(unsigned* __restrict__ regmax_u,
                             const float* __restrict__ h2c,
                             const int* __restrict__ counts, int D, int S) {
  const int i = blockIdx.x * 256 + threadIdx.x;  // B*NREG*D
  const int d = i & (D - 1);
  const int br = i / D;
  float f = __uint_as_float(regmax_u[i]);
  if (counts[br] < S) f = fmaxf(f, h2c[d]);       // masked rows contribute h_const
  ((float*)regmax_u)[i] = f;
}

// --------------------------- 7) loss ---------------------------------------
__global__ void loss_kernel(const float* __restrict__ sumlogit,
                            float* __restrict__ out_loss, int B, int S) {
  __shared__ float sh[128];
  const int t = threadIdx.x;
  float v = 0.f;
  if (t < B * NREG) {
    float s = sumlogit[t] / (float)S;
    v = s * s;
  }
  sh[t] = v;
  __syncthreads();
  for (int off = 64; off > 0; off >>= 1) {
    if (t < off) sh[t] += sh[t + off];
    __syncthreads();
  }
  if (t == 0) *out_loss = sh[0] / (float)B;
}

// --------------------------- 8) output assembly ----------------------------
// out[b,s] = [ x[b,s] | regmax_f[b, idx[b,s]] | g_vec[b] ]
__global__ __launch_bounds__(256) void out_kernel(
    const float* __restrict__ x, const float* __restrict__ g_vec,
    const float* __restrict__ regmax_f, const int* __restrict__ reg_indx,
    float* __restrict__ out, int S, int D) {
  const int row = blockIdx.x;
  const int b = row / S;
  const float* xr = x + (size_t)row * D;
  const float* rv = regmax_f + ((size_t)b * NREG + reg_indx[row]) * D;
  const float* gv = g_vec + (size_t)b * D;
  float* o = out + (size_t)row * 3 * D;
  for (int j = threadIdx.x; j < D; j += 256) {
    o[j]         = xr[j];
    o[D + j]     = rv[j];
    o[2 * D + j] = gv[j];
  }
}

// ---------------------------------------------------------------------------
extern "C" void kernel_launch(void* const* d_in, const int* in_sizes, int n_in,
                              void* d_out, int out_size, void* d_ws, size_t ws_size,
                              hipStream_t stream) {
  const float* x     = (const float*)d_in[0];
  const float* g_vec = (const float*)d_in[1];
  const float* occ_w = (const float*)d_in[2];
  const float* occ_b = (const float*)d_in[3];
  const float* w1    = (const float*)d_in[4];
  const float* b1    = (const float*)d_in[5];
  const float* w2    = (const float*)d_in[6];
  const float* b2    = (const float*)d_in[7];
  float* out = (float*)d_out;

  const int D    = in_sizes[2] / NREG;        // 512
  const int B    = in_sizes[1] / D;           // 8
  const int S    = in_sizes[0] / (B * D);     // 4096
  const int Mtot = B * S;                     // 32768

  char* ws = (char*)d_ws;
  size_t off = 0;
  unsigned short* h1 = (unsigned short*)(ws + off); off += (size_t)Mtot * D * sizeof(unsigned short);
  int*   reg_indx    = (int*)(ws + off);            off += (size_t)Mtot * sizeof(int);
  float* reg_logit   = (float*)(ws + off);          off += (size_t)Mtot * sizeof(float);
  size_t regmax_off  = off;
  unsigned* regmax   = (unsigned*)(ws + off);       off += (size_t)B * NREG * D * sizeof(unsigned);
  float* sumlogit    = (float*)(ws + off);          off += (size_t)B * NREG * sizeof(float);
  int*   counts      = (int*)(ws + off);            off += (size_t)B * NREG * sizeof(int);
  float* h2c         = (float*)(ws + off);          off += (size_t)D * sizeof(float);
  (void)ws_size; (void)n_in;

  (void)hipMemsetAsync(ws + regmax_off, 0, (size_t)B * NREG * D * sizeof(unsigned), stream);

  occ_kernel<<<Mtot / 8, 256, 0, stream>>>(x, occ_w, occ_b, reg_indx, reg_logit, S, D);
  stats_kernel<<<1, B * NREG, 0, stream>>>(reg_indx, reg_logit, sumlogit, counts, S);

  dim3 gg(D / BN, Mtot / BM);
  gemm1_kernel<<<gg, 256, 0, stream>>>(x, w1, b1, h1, D, D);
  gemm2_kernel<<<gg, 256, 0, stream>>>(h1, w2, b2, reg_indx, regmax, D, D, S);

  hconst_kernel<<<(D + 255) / 256, 256, 0, stream>>>(b1, w2, b2, h2c, D);
  merge_kernel<<<(B * NREG * D) / 256, 256, 0, stream>>>(regmax, h2c, counts, D, S);
  loss_kernel<<<1, 128, 0, stream>>>(sumlogit, out + (size_t)out_size - 1, B, S);
  out_kernel<<<Mtot, 256, 0, stream>>>(x, g_vec, (const float*)regmax, reg_indx, out, S, D);
}
